// RNN_16346645529381
// MI455X (gfx1250) — compile-verified, matrix operand order checked by reading, not run on hardware
//
#include <hip/hip_runtime.h>
#include <hip/hip_bf16.h>

#define VOCAB  37
#define EMB    256
#define HID    512
#define NLAYER 3
#define BATCH  512
#define SEQ    128

typedef __attribute__((ext_vector_type(16))) __bf16        v16bf;
typedef __attribute__((ext_vector_type(8)))  float         v8f;
typedef __attribute__((ext_vector_type(4)))  unsigned int  u32x4;
typedef __attribute__((ext_vector_type(8)))  int           i32x8;
typedef __attribute__((ext_vector_type(4)))  int           i32x4;

union FragU { v16bf v; uint4 q[2]; };

// Load a 16x32 bf16 fragment (A/B operand) per CDNA5 WMMA layout:
// lane r = lane&15 holds row (row_base + r); lane[4] selects K-halves:
//   lanes 0-15 : K = k0+0..7   and k0+16..23
//   lanes 16-31: K = k0+8..15  and k0+24..31
__device__ __forceinline__ v16bf load_frag_g(const __hip_bfloat16* __restrict__ base,
                                             int row_stride, int row_base,
                                             int k0, int lane) {
  const int r  = lane & 15;
  const int kb = (lane >> 4) << 3;
  const __hip_bfloat16* p = base + (size_t)(row_base + r) * row_stride + k0 + kb;
  FragU f;
  f.q[0] = *(const uint4*)(p);
  f.q[1] = *(const uint4*)(p + 16);
  return f.v;
}

// Same fragment fetch but from LDS (ds_load_b128 path).
__device__ __forceinline__ v16bf load_frag_lds(const __hip_bfloat16* base,
                                               int row_stride, int row_base,
                                               int k0, int lane) {
  const int r  = lane & 15;
  const int kb = (lane >> 4) << 3;
  const __hip_bfloat16* p = base + (row_base + r) * row_stride + k0 + kb;
  FragU f;
  f.q[0] = *(const uint4*)(p);
  f.q[1] = *(const uint4*)(p + 16);
  return f.v;
}

__device__ __forceinline__ float sigm_(float x)     { return 1.0f / (1.0f + __expf(-x)); }
__device__ __forceinline__ float tanh_fast(float x) { return 2.0f / (1.0f + __expf(-2.0f * x)) - 1.0f; }

// ---- TDM: load a 64-row weight panel (4 gate blocks x 16 rows x `width` K)
// into LDS as one 3D tensor tile. D# built per ISA 8.3-8.6:
//   dim0 = width (contiguous), dim1 = 16 rows (stride row_stride),
//   dim2 = 4 gate blocks (stride slice_stride), data_size = 2 bytes.
// This toolchain's builtin is the 6-arg form:
//   (u32x4 g0, i32x8 g1, i32x4 g2, i32x4 g3, i32x8 g4, i32 cpol)
__device__ __forceinline__ void tdm_load_panel(const __hip_bfloat16* gbase,
                                               unsigned lds_byte_off,
                                               int width,
                                               unsigned long long row_stride,
                                               unsigned long long slice_stride) {
  const unsigned long long ga = (unsigned long long)gbase;
  const unsigned td0 = (unsigned)width;   // tensor_dim0 (OOB bound)
  const unsigned td1 = 16u;               // tensor_dim1
  const unsigned long long s0 = row_stride;    // tensor_dim0_stride (48b)
  const unsigned long long s1 = slice_stride;  // tensor_dim1_stride (48b)

  u32x4 g0;
  g0[0] = 1u;                                              // count=1, user mode
  g0[1] = lds_byte_off;                                    // lds_addr
  g0[2] = (unsigned)(ga & 0xFFFFFFFFu);                    // global_addr lo
  g0[3] = (unsigned)((ga >> 32) & 0x01FFFFFFu) | (2u << 30);  // addr hi | type=2

  i32x8 g1;
  g1[0] = (int)(1u << 16);                                 // data_size=1 (2B), mask=0
  g1[1] = (int)((td0 & 0xFFFFu) << 16);                    // tensor_dim0[15:0]
  g1[2] = (int)(((td0 >> 16) & 0xFFFFu) | ((td1 & 0xFFFFu) << 16));
  g1[3] = (int)(((td1 >> 16) & 0xFFFFu) | ((unsigned)width << 16));  // tile_dim0
  g1[4] = (int)(16u | (4u << 16));                         // tile_dim1=16, tile_dim2=4
  g1[5] = (int)(unsigned)(s0 & 0xFFFFFFFFu);               // dim0_stride[31:0]
  g1[6] = (int)(((unsigned)(s0 >> 32) & 0xFFFFu) |
                (((unsigned)(s1 & 0xFFFFu)) << 16));       // dim0_stride[47:32] | dim1_stride[15:0]
  g1[7] = (int)(unsigned)((s1 >> 16) & 0xFFFFFFFFu);       // dim1_stride[47:16]

  i32x4 g2;
  g2[0] = 4;                                               // tensor_dim2 = 4 gate blocks
  g2[1] = 0; g2[2] = 0; g2[3] = 0;                         // no dim3, no iterate

  i32x4 g3; g3[0] = 0; g3[1] = 0; g3[2] = 0; g3[3] = 0;

  i32x8 g4;
  g4[0] = 0; g4[1] = 0; g4[2] = 0; g4[3] = 0;
  g4[4] = 0; g4[5] = 0; g4[6] = 0; g4[7] = 0;

  __builtin_amdgcn_tensor_load_to_lds(g0, g1, g2, g3, g4, 0);
}

// ---------------- elementwise helpers ----------------

__global__ void f32_to_bf16_kernel(const float* __restrict__ src,
                                   __hip_bfloat16* __restrict__ dst, int n) {
  int i = blockIdx.x * blockDim.x + threadIdx.x;
  if (i < n) dst[i] = __float2bfloat16(src[i]);
}

__global__ void embed_kernel(const int* __restrict__ tokens,
                             const float* __restrict__ emb,
                             __hip_bfloat16* __restrict__ X0) {
  const int bs = blockIdx.x;      // b*SEQ + s
  const int e  = threadIdx.x;     // 0..EMB-1
  const int tok = tokens[bs];
  X0[(size_t)bs * EMB + e] = __float2bfloat16(emb[(size_t)tok * EMB + e]);
}

// ---------------- fused LSTM step (TDM -> LDS -> WMMA) ----------------
// grid = (BATCH/64, HID/16), block = 128 (4 waves).
// Wave 0 issues two TENSOR_LOAD_TO_LDS ops pulling the workgroup's 64-row
// W_ih / W_hh panels (4 gates x 16 hidden cols, full K) into LDS, waits on
// TENSORcnt, then the workgroup barrier publishes the panel. All 4 waves then
// stream B-fragments from LDS and per-wave A-fragments (batch tiles) from
// global. gates = x_t @ W_ih^T + h_{t-1} @ W_hh^T (+ biases), fused update.
__global__ void __launch_bounds__(128)
lstm_step_kernel(const __hip_bfloat16* __restrict__ Xin, int Din,
                 const __hip_bfloat16* __restrict__ Wih,  // [4H, Din] bf16
                 const __hip_bfloat16* __restrict__ Whh,  // [4H, H]   bf16
                 const float* __restrict__ bih, const float* __restrict__ bhh,
                 __hip_bfloat16* __restrict__ Xout,       // [B, S, H] bf16 (h history)
                 float* __restrict__ Cst,                 // [B, H] f32 cell state
                 int t) {
  extern __shared__ __hip_bfloat16 smem[];   // [64*Din] Wih panel | [64*HID] Whh panel

  const int tid        = threadIdx.x;
  const int lane       = tid & 31;
  const int wave       = tid >> 5;
  const int batch_base = blockIdx.x * 64 + wave * 16;
  const int hid_base   = blockIdx.y * 16;

  __hip_bfloat16* sWih = smem;
  __hip_bfloat16* sWhh = smem + 64 * Din;

  // ---- TDM staging of both weight panels (wave 0 only; TDM ignores EXEC) ----
  if (wave == 0) {
    tdm_load_panel(Wih + (size_t)hid_base * Din, 0u, Din,
                   (unsigned long long)Din, (unsigned long long)HID * Din);
    tdm_load_panel(Whh + (size_t)hid_base * HID, (unsigned)(64 * Din * 2), HID,
                   (unsigned long long)HID, (unsigned long long)HID * HID);
    __builtin_amdgcn_s_wait_tensorcnt(0);
  }
  __syncthreads();

  v8f acc[4] = {};   // one 16x16 f32 accumulator per gate (i,f,g,o)

  // ---- x_t @ W_ih^T contribution ----
  const __hip_bfloat16* Abase = Xin + (size_t)t * Din;     // row b => +b*SEQ*Din
  for (int k0 = 0; k0 < Din; k0 += 32) {
    v16bf a  = load_frag_g(Abase, SEQ * Din, batch_base, k0, lane);
    v16bf b0 = load_frag_lds(sWih, Din,  0, k0, lane);
    v16bf b1 = load_frag_lds(sWih, Din, 16, k0, lane);
    v16bf b2 = load_frag_lds(sWih, Din, 32, k0, lane);
    v16bf b3 = load_frag_lds(sWih, Din, 48, k0, lane);
    acc[0] = __builtin_amdgcn_wmma_f32_16x16x32_bf16(false, a, false, b0, (short)0, acc[0], false, false);
    acc[1] = __builtin_amdgcn_wmma_f32_16x16x32_bf16(false, a, false, b1, (short)0, acc[1], false, false);
    acc[2] = __builtin_amdgcn_wmma_f32_16x16x32_bf16(false, a, false, b2, (short)0, acc[2], false, false);
    acc[3] = __builtin_amdgcn_wmma_f32_16x16x32_bf16(false, a, false, b3, (short)0, acc[3], false, false);
  }

  // ---- h_{t-1} @ W_hh^T contribution (h_0 == 0 -> skip) ----
  if (t > 0) {
    const __hip_bfloat16* Hbase = Xout + (size_t)(t - 1) * HID;  // row b => +b*SEQ*H
    for (int k0 = 0; k0 < HID; k0 += 32) {
      v16bf a  = load_frag_g(Hbase, SEQ * HID, batch_base, k0, lane);
      v16bf b0 = load_frag_lds(sWhh, HID,  0, k0, lane);
      v16bf b1 = load_frag_lds(sWhh, HID, 16, k0, lane);
      v16bf b2 = load_frag_lds(sWhh, HID, 32, k0, lane);
      v16bf b3 = load_frag_lds(sWhh, HID, 48, k0, lane);
      acc[0] = __builtin_amdgcn_wmma_f32_16x16x32_bf16(false, a, false, b0, (short)0, acc[0], false, false);
      acc[1] = __builtin_amdgcn_wmma_f32_16x16x32_bf16(false, a, false, b1, (short)0, acc[1], false, false);
      acc[2] = __builtin_amdgcn_wmma_f32_16x16x32_bf16(false, a, false, b2, (short)0, acc[2], false, false);
      acc[3] = __builtin_amdgcn_wmma_f32_16x16x32_bf16(false, a, false, b3, (short)0, acc[3], false, false);
    }
  }

  // ---- gate nonlinearities + state update ----
  // C/D layout: lanes 0-15: M=r, N=lane; lanes 16-31: M=8+r, N=lane-16.
  const int n  = hid_base + (lane & 15);
  const int mb = (lane >> 4) << 3;

  float bias[4];
#pragma unroll
  for (int g = 0; g < 4; ++g) bias[g] = bih[g * HID + n] + bhh[g * HID + n];

#pragma unroll
  for (int r = 0; r < 8; ++r) {
    const int b = batch_base + mb + r;
    const float gi = acc[0][r] + bias[0];
    const float gf = acc[1][r] + bias[1];
    const float gg = acc[2][r] + bias[2];
    const float go = acc[3][r] + bias[3];
    const float si = sigm_(gi);
    const float sf = sigm_(gf);
    const float sg = tanh_fast(gg);
    const float so = sigm_(go);
    const float cp = (t > 0) ? Cst[(size_t)b * HID + n] : 0.0f;
    const float cn = sf * cp + si * sg;
    Cst[(size_t)b * HID + n] = cn;
    Xout[((size_t)b * SEQ + t) * HID + n] = __float2bfloat16(so * tanh_fast(cn));
  }
}

// ---------------- hT / cT extraction ----------------
__global__ void extract_kernel(const __hip_bfloat16* __restrict__ Xout,
                               const float* __restrict__ Cst,
                               float* __restrict__ hT, float* __restrict__ cT) {
  const int i = blockIdx.x * blockDim.x + threadIdx.x;   // over B*H
  const int b = i / HID, n = i % HID;
  hT[i] = __bfloat162float(Xout[((size_t)b * SEQ + (SEQ - 1)) * HID + n]);
  cT[i] = Cst[i];
}

// ---------------- logits (V=37, tiny: plain VALU dot) ----------------
__global__ void logits_kernel(const __hip_bfloat16* __restrict__ X,   // [B,S,H] bf16
                              const float* __restrict__ W,            // [V,H]
                              const float* __restrict__ bo,
                              float* __restrict__ out) {              // [B,S,V]
  const long long idx   = (long long)blockIdx.x * blockDim.x + threadIdx.x;
  const long long total = (long long)BATCH * SEQ * VOCAB;
  if (idx >= total) return;
  const int       v  = (int)(idx % VOCAB);
  const long long bs = idx / VOCAB;
  const __hip_bfloat16* x = X + bs * HID;
  const float*          w = W + (size_t)v * HID;
  float acc = 0.0f;
#pragma unroll 8
  for (int k = 0; k < HID; ++k) acc += __bfloat162float(x[k]) * w[k];
  out[idx] = acc + bo[v];
}

// ---------------- host launcher ----------------
extern "C" void kernel_launch(void* const* d_in, const int* in_sizes, int n_in,
                              void* d_out, int out_size, void* d_ws, size_t ws_size,
                              hipStream_t stream) {
  (void)in_sizes; (void)n_in; (void)out_size; (void)ws_size;

  const int*   tokens  = (const int*)d_in[0];
  const float* emb     = (const float*)d_in[1];
  const float* W_ih[3] = {(const float*)d_in[2], (const float*)d_in[6],  (const float*)d_in[10]};
  const float* W_hh[3] = {(const float*)d_in[3], (const float*)d_in[7],  (const float*)d_in[11]};
  const float* b_ih[3] = {(const float*)d_in[4], (const float*)d_in[8],  (const float*)d_in[12]};
  const float* b_hh[3] = {(const float*)d_in[5], (const float*)d_in[9],  (const float*)d_in[13]};
  const float* W_out   = (const float*)d_in[14];
  const float* b_out   = (const float*)d_in[15];
  float* out = (float*)d_out;

  // workspace carve-up (256B aligned)
  char*  ws  = (char*)d_ws;
  size_t off = 0;
  auto carve = [&](size_t bytes) -> char* {
    char* p = ws + off;
    off += (bytes + 255) & ~(size_t)255;
    return p;
  };
  __hip_bfloat16* Xa = (__hip_bfloat16*)carve((size_t)BATCH * SEQ * HID * 2);  // 64 MB
  __hip_bfloat16* Xb = (__hip_bfloat16*)carve((size_t)BATCH * SEQ * HID * 2);  // 64 MB
  __hip_bfloat16* WihB[3];
  __hip_bfloat16* WhhB[3];
  const int din_l[3] = {EMB, HID, HID};
  for (int l = 0; l < NLAYER; ++l) {
    WihB[l] = (__hip_bfloat16*)carve((size_t)4 * HID * din_l[l] * 2);
    WhhB[l] = (__hip_bfloat16*)carve((size_t)4 * HID * HID * 2);
  }
  float* Cst = (float*)carve((size_t)BATCH * HID * 4);

  // 1) weights -> bf16
  for (int l = 0; l < NLAYER; ++l) {
    int nih = 4 * HID * din_l[l];
    int nhh = 4 * HID * HID;
    f32_to_bf16_kernel<<<(nih + 255) / 256, 256, 0, stream>>>(W_ih[l], WihB[l], nih);
    f32_to_bf16_kernel<<<(nhh + 255) / 256, 256, 0, stream>>>(W_hh[l], WhhB[l], nhh);
  }

  // 2) embedding gather -> bf16 layer-0 input (packed stride EMB)
  embed_kernel<<<BATCH * SEQ, EMB, 0, stream>>>(tokens, emb, Xa);

  // 3) layers: sequential WMMA steps + hT/cT extraction
  const size_t logitsN = (size_t)BATCH * SEQ * VOCAB;
  __hip_bfloat16* bufs[2] = {Xa, Xb};
  for (int l = 0; l < NLAYER; ++l) {
    __hip_bfloat16* Xin  = bufs[l & 1];
    __hip_bfloat16* Xout = bufs[(l + 1) & 1];
    const size_t smemB = (size_t)64 * (din_l[l] + HID) * 2;   // 96 KB / 128 KB
    for (int t = 0; t < SEQ; ++t) {
      lstm_step_kernel<<<dim3(BATCH / 64, HID / 16), 128, smemB, stream>>>(
          Xin, din_l[l], WihB[l], WhhB[l], b_ih[l], b_hh[l], Xout, Cst, t);
    }
    extract_kernel<<<(BATCH * HID) / 256, 256, 0, stream>>>(
        Xout, Cst,
        out + logitsN + (size_t)l * BATCH * HID,
        out + logitsN + (size_t)NLAYER * BATCH * HID + (size_t)l * BATCH * HID);
  }

  // 4) logits from final layer output (bufs[1] after 3 layers)
  const long long tot = (long long)BATCH * SEQ * VOCAB;
  logits_kernel<<<(unsigned)((tot + 255) / 256), 256, 0, stream>>>(bufs[1], W_out, b_out, out);
}
